// SpatialNeuralProcess_49237505081885
// MI455X (gfx1250) — matrix operand order repaired and checked by reading, not compiled, not run on hardware
//
#include <hip/hip_runtime.h>

// ---------------------------------------------------------------------------
// Spatial Neural Process forward for MI455X (gfx1250, wave32, WMMA).
// All dense contractions use v_wmma_f32_16x16x32_bf16 fed by pre-staged bf16
// operands (one conversion pass per buffer, contiguous b128 loads in-loop).
// Attention is fused flash-style per wave (16 q-rows x 1 head), computing
// S^T = K*Q^T so the softmax recurrence is per-lane and P lands directly in
// the A-operand layout for the P*V WMMA.
// ---------------------------------------------------------------------------

typedef __bf16 v16bf __attribute__((ext_vector_type(16)));
typedef float  v8f   __attribute__((ext_vector_type(8)));

union BF16x16 { v16bf v; unsigned short s[16]; unsigned u[8]; };

__device__ __forceinline__ unsigned short f2bf(float f) {
  union { float f; unsigned u; } c; c.f = f;
  unsigned u = c.u;
  u += 0x7fffu + ((u >> 16) & 1u);   // round-to-nearest-even
  return (unsigned short)(u >> 16);
}

// pack two floats into one dword of bf16 (lo -> [15:0], hi -> [31:16])
__device__ __forceinline__ unsigned f2bf_pk(float lo, float hi) {
  return (unsigned)f2bf(lo) | ((unsigned)f2bf(hi) << 16);
}

// load 4 packed-bf16 dwords (8 bf16) as one b128 into u[o..o+3]
__device__ __forceinline__ void ld8(BF16x16& d, int o, const unsigned short* p) {
  uint4 x = *reinterpret_cast<const uint4*>(p);
  d.u[o + 0] = x.x; d.u[o + 1] = x.y; d.u[o + 2] = x.z; d.u[o + 3] = x.w;
}

__device__ __forceinline__ v8f wmma_bf16(v16bf a, v16bf b, v8f c) {
  return __builtin_amdgcn_wmma_f32_16x16x32_bf16(
      /*neg_a=*/false, a, /*neg_b=*/false, b,
      /*c_mod=*/(short)0, c, /*reuse_a=*/false, /*reuse_b=*/false);
}

__global__ void fill_zero(float* __restrict__ p, int n) {
  int i = blockIdx.x * blockDim.x + threadIdx.x;
  if (i < n) p[i] = 0.f;
}

// f32 -> bf16 (row-major copy), n even, 2 elements per thread
__global__ void cvt_bf16(const float* __restrict__ in,
                         unsigned short* __restrict__ out, int n) {
  int i = blockIdx.x * blockDim.x + threadIdx.x;
  int i2 = i * 2;
  if (i2 < n) ((unsigned*)out)[i] = f2bf_pk(in[i2], in[i2 + 1]);
}

// in[b][R][Cc] (f32) -> out[b][Cc][R] (bf16); R even; grid.y = batch.
// Used for weight transposes ([K][128] -> [128][K]) and V^T staging.
__global__ void transpose_cvt_bf16(const float* __restrict__ in,
                                   unsigned short* __restrict__ out,
                                   int R, int Cc, long iStride, long oStride) {
  long idx = (long)blockIdx.x * blockDim.x + threadIdx.x;
  long total = (long)(R >> 1) * Cc;
  if (idx >= total) return;
  int b = blockIdx.y;
  int rh = (int)(idx % (R >> 1));
  int c  = (int)(idx / (R >> 1));
  int r = rh * 2;
  const float* ib = in + (long)b * iStride;
  unsigned v = f2bf_pk(ib[(long)r * Cc + c], ib[(long)(r + 1) * Cc + c]);
  ((unsigned*)out)[((long)b * oStride + (long)c * R + r) >> 1] = v;
}

// ---------------------------------------------------------------------------
// GEMM, N fixed at 128: C[M,128] = act(A[M,K]bf16 @ Wt[128,K]bf16 + bias (+resid))
// 256 threads = 8 waves; wave tile 16(M) x 64(N); WG tile 64 x 128.
// grid = (M/64, 1, batches). Strides in elements (0 for shared operands).
// bias must always be a valid pointer (use a zero vector for "no bias").
// ---------------------------------------------------------------------------
__global__ __launch_bounds__(256) void gemm_wmma_n128(
    const unsigned short* __restrict__ A, long aStride,
    const unsigned short* __restrict__ Wt, long wStride,
    const float* __restrict__ bias,
    const float* __restrict__ resid, long rStride,
    float* __restrict__ C, long cStride,
    int M, int K, int relu) {
  int lane = threadIdx.x & 31;
  int wave = threadIdx.x >> 5;
  long b = blockIdx.z;
  A += b * aStride; Wt += b * wStride; C += b * cStride;

  int m0 = blockIdx.x * 64 + (wave >> 1) * 16;
  int n0 = (wave & 1) * 64;
  int half = lane >> 4, l15 = lane & 15;

  const unsigned short* arow = A + (long)(m0 + l15) * K + half * 8;
  v8f acc[4] = {};
  for (int k0 = 0; k0 < K; k0 += 32) {
    BF16x16 a;
    ld8(a, 0, arow + k0);        // K = k0+half*8   .. +7
    ld8(a, 4, arow + k0 + 16);   // K = k0+16+half*8.. +7
#pragma unroll
    for (int t = 0; t < 4; ++t) {
      BF16x16 bm;
      const unsigned short* wp =
          Wt + (long)(n0 + t * 16 + l15) * K + k0 + half * 8;
      ld8(bm, 0, wp);
      ld8(bm, 4, wp + 16);
      acc[t] = wmma_bf16(a.v, bm.v, acc[t]);
    }
  }

  const float lo = relu ? 0.f : -3.402823466e38f;
  if (resid) {
    const float* R = resid + b * rStride;
#pragma unroll
    for (int t = 0; t < 4; ++t) {
      int col = n0 + t * 16 + l15;
      float bval = bias[col];
#pragma unroll
      for (int v = 0; v < 8; ++v) {
        int row = m0 + half * 8 + v;
        float val = acc[t][v] + bval + R[(long)row * 128 + col];
        C[(long)row * 128 + col] = fmaxf(val, lo);
      }
    }
  } else {
#pragma unroll
    for (int t = 0; t < 4; ++t) {
      int col = n0 + t * 16 + l15;
      float bval = bias[col];
#pragma unroll
      for (int v = 0; v < 8; ++v) {
        int row = m0 + half * 8 + v;
        float val = acc[t][v] + bval;
        C[(long)row * 128 + col] = fmaxf(val, lo);
      }
    }
  }
}

// ---------------------------------------------------------------------------
// Fused flash attention, d=32 per head, 512 keys, 4 heads, B=4.
// Q,K: bf16 row-major [B*512,128]. Vt: bf16 [B][128 d][512 keys].
// One wave handles (b, h, 16 q-rows). grid = 128 blocks x 128 threads.
// ---------------------------------------------------------------------------
__global__ __launch_bounds__(128) void flash_attn(
    const unsigned short* __restrict__ Q, const unsigned short* __restrict__ K,
    const unsigned short* __restrict__ Vt, float* __restrict__ out) {
  const int NKEY = 512;
  int lane = threadIdx.x & 31;
  int wave = threadIdx.x >> 5;
  int wg = blockIdx.x * 4 + wave;        // 0..511
  int qt = wg & 31;                      // q tile (16 rows)
  int h  = (wg >> 5) & 3;                // head
  int b  = wg >> 7;                      // batch
  int qbase = qt * 16;
  const unsigned short* Qb = Q + (long)b * 512 * 128 + h * 32;
  const unsigned short* Kb = K + (long)b * 512 * 128 + h * 32;
  const unsigned short* Vb = Vt + (long)b * 128 * 512 + (long)(h * 32) * 512;
  float* Ob = out + (long)b * 512 * 128 + h * 32;

  int half = lane >> 4, l15 = lane & 15;
  const float scale = 0.17677669529663687f;  // 1/sqrt(32)

  // Q^T as B-operand (32x16): lane = q-row column, half h holds d 8h..8h+7 and 16+8h..16+8h+7.
  BF16x16 qb;
  {
    const unsigned short* qr = Qb + (long)(qbase + l15) * 128 + half * 8;
    ld8(qb, 0, qr);
    ld8(qb, 4, qr + 16);
  }

  // ---- pass 1: per-lane online softmax stats over this lane's key half ----
  float m = -1e30f, lsum = 0.f;
  for (int c0 = 0; c0 < NKEY; c0 += 16) {
    BF16x16 ak;
    const unsigned short* kr = Kb + (long)(c0 + l15) * 128 + half * 8;
    ld8(ak, 0, kr);
    ld8(ak, 4, kr + 16);
    v8f z = {};
    v8f s = wmma_bf16(ak.v, qb.v, z);
#pragma unroll
    for (int v = 0; v < 8; ++v) {
      float sv = s[v] * scale;
      float mn = fmaxf(m, sv);
      lsum = lsum * __expf(m - mn) + __expf(sv - mn);
      m = mn;
    }
  }
  // merge the two key-halves (lanes l and l+16 hold the same q-row)
  float mo = __shfl_xor(m, 16, 32);
  float lo = __shfl_xor(lsum, 16, 32);
  float mt = fmaxf(m, mo);
  float lt = lsum * __expf(m - mt) + lo * __expf(mo - mt);

  __shared__ float srow_l[4][16];
  if (half == 0) srow_l[wave][l15] = lt;
  __syncthreads();

  // ---- pass 2: recompute scores, exp, accumulate P*V ----
  v8f acc0 = {}, acc1 = {};
  for (int c0 = 0; c0 < NKEY; c0 += 32) {
    v8f s0, s1;
    {
      BF16x16 ak;
      const unsigned short* kr = Kb + (long)(c0 + l15) * 128 + half * 8;
      ld8(ak, 0, kr);
      ld8(ak, 4, kr + 16);
      v8f z = {};
      s0 = wmma_bf16(ak.v, qb.v, z);
    }
    {
      BF16x16 ak;
      const unsigned short* kr = Kb + (long)(c0 + 16 + l15) * 128 + half * 8;
      ld8(ak, 0, kr);
      ld8(ak, 4, kr + 16);
      v8f z = {};
      s1 = wmma_bf16(ak.v, qb.v, z);
    }
    // P in A-operand layout: tile c0 -> s[0..7], tile c0+16 -> s[8..15]
    BF16x16 ap;
#pragma unroll
    for (int j = 0; j < 4; ++j) {
      ap.u[j]     = f2bf_pk(__expf(s0[2 * j] * scale - mt),
                            __expf(s0[2 * j + 1] * scale - mt));
      ap.u[4 + j] = f2bf_pk(__expf(s1[2 * j] * scale - mt),
                            __expf(s1[2 * j + 1] * scale - mt));
    }
    // V^T as B-operand: per lane keys krow0..+7 and krow0+16..+23 for one d column
    int krow0 = c0 + half * 8;
    BF16x16 bv0, bv1;
    {
      const unsigned short* v0p = Vb + (long)l15 * 512 + krow0;
      const unsigned short* v1p = Vb + (long)(16 + l15) * 512 + krow0;
      ld8(bv0, 0, v0p); ld8(bv0, 4, v0p + 16);
      ld8(bv1, 0, v1p); ld8(bv1, 4, v1p + 16);
    }
    acc0 = wmma_bf16(ap.v, bv0.v, acc0);
    acc1 = wmma_bf16(ap.v, bv1.v, acc1);
  }

  // epilogue: normalize by per-row sumexp and store
  int rbase = half * 8;
#pragma unroll
  for (int v = 0; v < 8; ++v) {
    int row = rbase + v;
    float inv = 1.f / srow_l[wave][row];
    float* orow = Ob + (long)(qbase + row) * 128;
    orow[l15]      = acc0[v] * inv;
    orow[16 + l15] = acc1[v] * inv;
  }
}

// ---------------------------------------------------------------------------
// Small-K input projection: out[m, outOff+n] = b[n] + sum_j srcs[m,j] W[j,n]
// ---------------------------------------------------------------------------
__global__ void small_linear(
    const float* __restrict__ s0, int s0s, int s0o, int n0,
    const float* __restrict__ s1, int s1s, int s1o, int n1,
    const float* __restrict__ W, const float* __restrict__ bias,
    float* __restrict__ out, int outStride, int outOff, int M, int N) {
  int idx = blockIdx.x * blockDim.x + threadIdx.x;
  if (idx >= M * N) return;
  int n = idx % N, m = idx / N;
  float acc = bias[n];
  for (int j = 0; j < n0; ++j) acc += s0[(long)m * s0s + s0o + j] * W[j * N + n];
  for (int j = 0; j < n1; ++j) acc += s1[(long)m * s1s + s1o + j] * W[(n0 + j) * N + n];
  out[(long)m * outStride + outOff + n] = acc;
}

__global__ void copy_cols(const float* __restrict__ src, int ss, int so,
                          float* __restrict__ dst, int ds, int dof,
                          int ncols, int rows) {
  int idx = blockIdx.x * blockDim.x + threadIdx.x;
  if (idx >= rows * ncols) return;
  int c = idx % ncols, r = idx / ncols;
  dst[(long)r * ds + dof + c] = src[(long)r * ss + so + c];
}

// LayerNorm over 128 cols; one wave per row, 4 waves per block.
__global__ __launch_bounds__(128) void layer_norm128(
    const float* __restrict__ x, const float* __restrict__ g,
    const float* __restrict__ b, float* __restrict__ out, int rows) {
  int lane = threadIdx.x & 31, wave = threadIdx.x >> 5;
  int row = blockIdx.x * 4 + wave;
  if (row >= rows) return;
  const float* xr = x + (long)row * 128;
  float v0 = xr[lane], v1 = xr[lane + 32], v2 = xr[lane + 64], v3 = xr[lane + 96];
  float s = v0 + v1 + v2 + v3;
  for (int o = 16; o; o >>= 1) s += __shfl_xor(s, o, 32);
  float mu = s * (1.f / 128.f);
  float d0 = v0 - mu, d1 = v1 - mu, d2 = v2 - mu, d3 = v3 - mu;
  float ss = d0 * d0 + d1 * d1 + d2 * d2 + d3 * d3;
  for (int o = 16; o; o >>= 1) ss += __shfl_xor(ss, o, 32);
  float rstd = rsqrtf(ss * (1.f / 128.f) + 1e-5f);
  float* orow = out + (long)row * 128;
  orow[lane]      = d0 * rstd * g[lane]      + b[lane];
  orow[lane + 32] = d1 * rstd * g[lane + 32] + b[lane + 32];
  orow[lane + 64] = d2 * rstd * g[lane + 64] + b[lane + 64];
  orow[lane + 96] = d3 * rstd * g[lane + 96] + b[lane + 96];
}

// key[b,c,h] -> keyT[b,h,c] (f32, coalesced writes)
__global__ void transpose_bch(const float* __restrict__ in, float* __restrict__ out) {
  int idx = blockIdx.x * blockDim.x + threadIdx.x;  // B*128*512
  if (idx >= 4 * 128 * 512) return;
  int c = idx & 511;
  int h = (idx >> 9) & 127;
  int b = idx >> 16;
  out[idx] = in[((long)b * 512 + c) * 128 + h];
}

// loc-encoder L1 distance logits + softmax over the 512 context points.
__global__ __launch_bounds__(256) void loc_w_softmax(
    const float* __restrict__ keyT, const float* __restrict__ query,
    float* __restrict__ wout) {
  int bt = blockIdx.x;
  int b = bt >> 9;
  __shared__ float q[128];
  __shared__ float redm[8];
  __shared__ float reds[8];
  int tid = threadIdx.x;
  if (tid < 128) q[tid] = query[(long)bt * 128 + tid];
  __syncthreads();
  const float* kb = keyT + (long)b * 128 * 512;
  int c0 = tid, c1 = tid + 256;
  float l0 = 0.f, l1 = 0.f;
  for (int h = 0; h < 128; ++h) {
    float qh = q[h];
    const float* kr = kb + (long)h * 512;
    l0 -= fabsf(0.5f * (kr[c0] - qh));
    l1 -= fabsf(0.5f * (kr[c1] - qh));
  }
  int lane = tid & 31, wave = tid >> 5;
  float mx = fmaxf(l0, l1);
  for (int o = 16; o; o >>= 1) mx = fmaxf(mx, __shfl_xor(mx, o, 32));
  if (lane == 0) redm[wave] = mx;
  __syncthreads();
  float bmx = redm[0];
  for (int i = 1; i < 8; ++i) bmx = fmaxf(bmx, redm[i]);
  float e0 = __expf(l0 - bmx), e1 = __expf(l1 - bmx);
  float sm = e0 + e1;
  for (int o = 16; o; o >>= 1) sm += __shfl_xor(sm, o, 32);
  if (lane == 0) reds[wave] = sm;
  __syncthreads();
  float bsum = 0.f;
  for (int i = 0; i < 8; ++i) bsum += reds[i];
  float inv = 1.f / bsum;
  float* wr = wout + (long)bt * 512;
  wr[c0] = e0 * inv;
  wr[c1] = e1 * inv;
}

// 128->1 head: mu (mode 0) or sigma = 0.1 + 0.9*softplus (mode 1).
__global__ __launch_bounds__(128) void dot_head(
    const float* __restrict__ H, const float* __restrict__ w,
    const float* __restrict__ bias, float* __restrict__ out, int rows, int mode) {
  int lane = threadIdx.x & 31, wave = threadIdx.x >> 5;
  int row = blockIdx.x * 4 + wave;
  if (row >= rows) return;
  const float* hr = H + (long)row * 128;
  float s = hr[lane] * w[lane] + hr[lane + 32] * w[lane + 32] +
            hr[lane + 64] * w[lane + 64] + hr[lane + 96] * w[lane + 96];
  for (int o = 16; o; o >>= 1) s += __shfl_xor(s, o, 32);
  if (lane == 0) {
    float v = s + bias[0];
    if (mode) {
      float sp = (v > 20.f) ? v : log1pf(__expf(v));
      v = 0.1f + 0.9f * sp;
    }
    out[row] = v;
  }
}

// ---------------------------------------------------------------------------
extern "C" void kernel_launch(void* const* d_in, const int* in_sizes, int n_in,
                              void* d_out, int out_size, void* d_ws, size_t ws_size,
                              hipStream_t stream) {
  (void)in_sizes; (void)n_in; (void)out_size; (void)ws_size;
  const float* cx = (const float*)d_in[0];  // [4,512,6]
  const float* cy = (const float*)d_in[1];  // [4,512,1]
  const float* tx = (const float*)d_in[2];  // [4,512,6]

  // JAX pytree flatten order: top keys sorted (dec,det,loc,var); lin={b,w};
  // attn={be,fin(b,w),g,wk,wq,wv}; encoder dicts sorted (attn,ctx,inp,tgt).
  const float* dec_attr_b = (const float*)d_in[3];
  const float* dec_attr_w = (const float*)d_in[4];
  const float* dec_d1a_b  = (const float*)d_in[5];
  const float* dec_d1a_w  = (const float*)d_in[6];
  const float* dec_d1b_b  = (const float*)d_in[7];
  const float* dec_d1b_w  = (const float*)d_in[8];
  const float* dec_d2a_b  = (const float*)d_in[9];
  const float* dec_d2a_w  = (const float*)d_in[10];
  const float* dec_d2b_b  = (const float*)d_in[11];
  const float* dec_d2b_w  = (const float*)d_in[12];
  const float* dec_loc_b  = (const float*)d_in[13];
  const float* dec_loc_w  = (const float*)d_in[14];

  struct AttnW { const float *be, *finb, *finw, *g, *wk, *wq, *wv; };
  auto load_attn = [&](int base) {
    AttnW a;
    a.be   = (const float*)d_in[base + 0];
    a.finb = (const float*)d_in[base + 1];
    a.finw = (const float*)d_in[base + 2];
    a.g    = (const float*)d_in[base + 3];
    a.wk   = (const float*)d_in[base + 4];
    a.wq   = (const float*)d_in[base + 5];
    a.wv   = (const float*)d_in[base + 6];
    return a;
  };
  AttnW det_a0 = load_attn(15), det_a1 = load_attn(22);
  const float* det_ctx_b = (const float*)d_in[29];
  const float* det_ctx_w = (const float*)d_in[30];
  const float* det_inp_b = (const float*)d_in[31];
  const float* det_inp_w = (const float*)d_in[32];
  const float* det_tgt_b = (const float*)d_in[33];
  const float* det_tgt_w = (const float*)d_in[34];
  const float* loc_ctx_b = (const float*)d_in[35];
  const float* loc_ctx_w = (const float*)d_in[36];
  const float* loc_inp_b = (const float*)d_in[37];
  const float* loc_inp_w = (const float*)d_in[38];
  const float* loc_tgt_b = (const float*)d_in[39];
  const float* loc_tgt_w = (const float*)d_in[40];
  AttnW var_a0 = load_attn(41), var_a1 = load_attn(48);
  const float* var_ctx_b = (const float*)d_in[55];
  const float* var_ctx_w = (const float*)d_in[56];
  const float* var_inp_b = (const float*)d_in[57];
  const float* var_inp_w = (const float*)d_in[58];
  const float* var_tgt_b = (const float*)d_in[59];
  const float* var_tgt_w = (const float*)d_in[60];

  // ---- workspace layout (floats), S = 4*512 = 2048 rows, H = 128 ----
  const long SH = 2048L * 128;
  float* ws = (float*)d_ws;
  long off = 0;
  auto alloc = [&](long n) { float* p = ws + off; off += n; return p; };
  float* det_ei = alloc(SH); float* det_k = alloc(SH); float* det_q = alloc(SH);
  float* var_ei = alloc(SH); float* var_k = alloc(SH); float* var_q = alloc(SH);
  float* loc_val = alloc(SH); float* loc_key = alloc(SH);
  float* loc_keyT = alloc(SH); float* loc_q = alloc(SH);
  float* wbuf = alloc(4L * 512 * 512);
  float* rep = alloc(SH); float* wout = alloc(SH);
  float* Kp = alloc(SH); float* Vp = alloc(SH); float* Qp = alloc(SH);
  float* attn_out = alloc(SH);
  float* cat = alloc(2 * SH);
  float* fin = alloc(SH);
  float* z1 = alloc(2048L * 320);
  float* z2 = alloc(2048L * 192);
  float* h1 = alloc(SH); float* h2 = alloc(SH);
  float* zbias = alloc(128);
  // bf16 staging (ushort counts halved into float units)
  unsigned short* abf  = (unsigned short*)alloc(524288);  // up to 4*512*512 bf16
  unsigned short* wtbf = (unsigned short*)alloc(131072);  // up to 4*128*512 bf16
  unsigned short* Qbf  = (unsigned short*)alloc(131072);
  unsigned short* Kbf  = (unsigned short*)alloc(131072);
  unsigned short* Vtbf = (unsigned short*)alloc(131072);

  dim3 gB(256);
  int nSH = (int)(SH);
  int cpBlocks = (nSH + 255) / 256;

  fill_zero<<<1, 128, 0, stream>>>(zbias, 128);

  auto cvt = [&](const float* src, unsigned short* dst, long n) {
    cvt_bf16<<<(int)((n / 2 + 255) / 256), 256, 0, stream>>>(src, dst, (int)n);
  };
  // W [K][128] f32 -> Wt [128][K] bf16 (single batch)
  auto wt = [&](const float* W, int K) {
    int thr = (K / 2) * 128;
    transpose_cvt_bf16<<<dim3((thr + 255) / 256, 1), 256, 0, stream>>>(
        W, wtbf, K, 128, 0, 0);
  };

  auto gemm = [&](const float* A, const float* W, const float* bias,
                  const float* resid, float* C, int M, int K, int relu) {
    cvt(A, abf, (long)M * K);
    wt(W, K);
    gemm_wmma_n128<<<dim3(M / 64, 1, 1), gB, 0, stream>>>(
        abf, 0, wtbf, 0, bias ? bias : zbias, resid, 0, C, 0, M, K, relu);
  };

  auto attention = [&](float* kbuf, float* vbuf, float* qbuf, const AttnW& a) {
    gemm(kbuf, a.wk, nullptr, nullptr, Kp, 2048, 128, 0);
    gemm(vbuf, a.wv, nullptr, nullptr, Vp, 2048, 128, 0);
    gemm(qbuf, a.wq, nullptr, nullptr, Qp, 2048, 128, 0);
    cvt(Qp, Qbf, SH);
    cvt(Kp, Kbf, SH);
    // Vp [b][512][128] -> Vt [b][128][512] bf16
    transpose_cvt_bf16<<<dim3(((512 / 2) * 128 + 255) / 256, 4), 256, 0, stream>>>(
        Vp, Vtbf, 512, 128, 512L * 128, 128L * 512);
    flash_attn<<<128, 128, 0, stream>>>(Qbf, Kbf, Vtbf, attn_out);
    copy_cols<<<cpBlocks, 256, 0, stream>>>(qbuf, 128, 0, cat, 256, 0, 128, 2048);
    copy_cols<<<cpBlocks, 256, 0, stream>>>(attn_out, 128, 0, cat, 256, 128, 128, 2048);
    gemm(cat, a.finw, a.finb, /*resid=*/qbuf, fin, 2048, 256, 0);
    layer_norm128<<<512, 128, 0, stream>>>(fin, a.g, a.be, qbuf, 2048);
  };

  int slBlocks128 = (2048 * 128 + 255) / 256;
  int slBlocks32  = (2048 * 32 + 255) / 256;

  // ---- det encoder ----
  small_linear<<<slBlocks128, 256, 0, stream>>>(cx, 6, 2, 4, cy, 1, 0, 1,
      det_inp_w, det_inp_b, det_ei, 128, 0, 2048, 128);
  small_linear<<<slBlocks128, 256, 0, stream>>>(tx, 6, 2, 4, nullptr, 0, 0, 0,
      det_tgt_w, det_tgt_b, det_q, 128, 0, 2048, 128);
  small_linear<<<slBlocks128, 256, 0, stream>>>(cx, 6, 2, 4, nullptr, 0, 0, 0,
      det_ctx_w, det_ctx_b, det_k, 128, 0, 2048, 128);
  attention(det_k, det_ei, det_q, det_a0);
  attention(det_k, det_ei, det_q, det_a1);

  // ---- var encoder ----
  small_linear<<<slBlocks128, 256, 0, stream>>>(cx, 6, 0, 6, nullptr, 0, 0, 0,
      var_inp_w, var_inp_b, var_ei, 128, 0, 2048, 128);
  small_linear<<<slBlocks128, 256, 0, stream>>>(tx, 6, 0, 6, nullptr, 0, 0, 0,
      var_tgt_w, var_tgt_b, var_q, 128, 0, 2048, 128);
  small_linear<<<slBlocks128, 256, 0, stream>>>(cx, 6, 0, 6, nullptr, 0, 0, 0,
      var_ctx_w, var_ctx_b, var_k, 128, 0, 2048, 128);
  attention(var_k, var_ei, var_q, var_a0);
  attention(var_k, var_ei, var_q, var_a1);

  // ---- loc encoder ----
  small_linear<<<slBlocks128, 256, 0, stream>>>(cx, 6, 0, 2, cy, 1, 0, 1,
      loc_inp_w, loc_inp_b, loc_val, 128, 0, 2048, 128);
  small_linear<<<slBlocks128, 256, 0, stream>>>(cx, 6, 0, 2, nullptr, 0, 0, 0,
      loc_ctx_w, loc_ctx_b, loc_key, 128, 0, 2048, 128);
  small_linear<<<slBlocks128, 256, 0, stream>>>(tx, 6, 0, 2, nullptr, 0, 0, 0,
      loc_ctx_w, loc_ctx_b, loc_q, 128, 0, 2048, 128);
  transpose_bch<<<(4 * 128 * 512 + 255) / 256, 256, 0, stream>>>(loc_key, loc_keyT);
  loc_w_softmax<<<2048, 256, 0, stream>>>(loc_keyT, loc_q, wbuf);
  // rep[b] = w[b](512x512) @ value[b](512x128): batched GEMM
  cvt(wbuf, abf, 4L * 512 * 512);
  transpose_cvt_bf16<<<dim3(((512 / 2) * 128 + 255) / 256, 4), 256, 0, stream>>>(
      loc_val, wtbf, 512, 128, 512L * 128, 128L * 512);
  gemm_wmma_n128<<<dim3(8, 1, 4), gB, 0, stream>>>(
      abf, 512L * 512, wtbf, 128L * 512, zbias, nullptr, 0,
      rep, 512L * 128, 512, 512, 0);
  gemm(rep, loc_tgt_w, loc_tgt_b, nullptr, wout, 2048, 128, 0);

  // ---- decoder ----
  // z1 = [w(128) | t_x(32) | s_x(32) | r(128)] (320); z2 = [t_x | s_x | v] (192)
  small_linear<<<slBlocks32, 256, 0, stream>>>(tx, 6, 2, 4, nullptr, 0, 0, 0,
      dec_attr_w, dec_attr_b, z1, 320, 128, 2048, 32);
  small_linear<<<slBlocks32, 256, 0, stream>>>(tx, 6, 0, 2, nullptr, 0, 0, 0,
      dec_loc_w, dec_loc_b, z1, 320, 160, 2048, 32);
  small_linear<<<slBlocks32, 256, 0, stream>>>(tx, 6, 2, 4, nullptr, 0, 0, 0,
      dec_attr_w, dec_attr_b, z2, 192, 0, 2048, 32);
  small_linear<<<slBlocks32, 256, 0, stream>>>(tx, 6, 0, 2, nullptr, 0, 0, 0,
      dec_loc_w, dec_loc_b, z2, 192, 32, 2048, 32);
  copy_cols<<<cpBlocks, 256, 0, stream>>>(wout, 128, 0, z1, 320, 0, 128, 2048);
  copy_cols<<<cpBlocks, 256, 0, stream>>>(det_q, 128, 0, z1, 320, 192, 128, 2048);
  copy_cols<<<cpBlocks, 256, 0, stream>>>(var_q, 128, 0, z2, 192, 64, 128, 2048);

  gemm(z1, dec_d1a_w, dec_d1a_b, nullptr, h1, 2048, 320, 1);
  dot_head<<<512, 128, 0, stream>>>(h1, dec_d1b_w, dec_d1b_b,
                                    (float*)d_out, 2048, 0);
  gemm(z2, dec_d2a_w, dec_d2a_b, nullptr, h2, 2048, 192, 1);
  dot_head<<<512, 128, 0, stream>>>(h2, dec_d2b_w, dec_d2b_b,
                                    (float*)d_out + 2048, 2048, 1);
}
